// GNNModel_60498909331789
// MI455X (gfx1250) — matrix operand order repaired and checked by reading, not compiled, not run on hardware
//
#include <hip/hip_runtime.h>
#include <hip/hip_bf16.h>
#include <math.h>

// ---------------------------------------------------------------------------
// GCN forward for MI455X (gfx1250, wave32).
//   h0 = relu(BN(x @ in_W))                (bias cancels inside BN)
//   for l in 0..5: h = relu(BN(scatter(h @ W_l))) + h
//   policy[n] = relu(h @ pW1 + pb1) @ pW2 + pb2
//   value     = tanh(relu(mean(h) @ vW1 + vb1) @ vW2 + vb2)
// GEMMs: v_wmma_f32_16x16x32_f16 (f16 A/B, f32 acc); weights staged per-block
// into LDS with GLOBAL_LOAD_ASYNC_TO_LDS_B128 (ASYNCcnt) when available.
// Scatter is warp-per-edge with f32 global atomics; the three Nx128 f32
// arrays (~154MB) mostly live in the 192MB L2.
// ---------------------------------------------------------------------------

typedef __attribute__((ext_vector_type(16))) _Float16 v16h;
typedef __attribute__((ext_vector_type(8)))  float    v8f;

// Builtin prototype (from hipcc diagnostic): global int4* src, LDS int4* dst.
typedef int v4i __attribute__((vector_size(16)));
typedef __attribute__((address_space(1))) v4i g_v4i;   // global
typedef __attribute__((address_space(3))) v4i l_v4i;   // LDS

#if defined(__AMDGCN__) && \
    __has_builtin(__builtin_amdgcn_global_load_async_to_lds_b128) && \
    __has_builtin(__builtin_amdgcn_s_wait_asynccnt)
#define USE_ASYNC_LDS 1
#else
#define USE_ASYNC_LDS 0
#endif

#define H 128
#define LAYERS 6
#define EPSV 1e-5f

// ---- degree / normalization ------------------------------------------------
__global__ void k_deg_init(float* __restrict__ deg, int n) {
    int i = blockIdx.x * blockDim.x + threadIdx.x;
    if (i < n) deg[i] = 1.0f;                       // self loop
}
__global__ void k_deg_edges(const int* __restrict__ dst, float* __restrict__ deg, int E) {
    int i = blockIdx.x * blockDim.x + threadIdx.x;
    if (i < E) atomicAdd(&deg[dst[i]], 1.0f);
}
__global__ void k_dinv(float* __restrict__ deg, int n) {
    int i = blockIdx.x * blockDim.x + threadIdx.x;
    if (i < n) deg[i] = rsqrtf(deg[i]);             // deg >= 1 always
}

// ---- pack conv weights into WMMA-B-fragment-native f16 layout --------------
// fragment (layer, kt, ct): 32 lanes x 16 halves, contiguous.
// lane l holds column n = ct*16 + (l&15); K range: lanes 0-15 -> kt*32+0..15,
// lanes 16-31 -> kt*32+16..31; element e: j=e/2, k = kbase + 2j + (e&1).
__global__ void k_pack_w(const float* __restrict__ convW, _Float16* __restrict__ Wp, int total) {
    int i = blockIdx.x * blockDim.x + threadIdx.x;
    if (i >= total) return;
    int e     = i & 15;
    int l     = (i >> 4) & 31;
    int ct    = (i >> 9) & 7;
    int kt    = (i >> 12) & 3;
    int layer = i >> 14;
    int j = e >> 1, hl = e & 1;
    int k = kt * 32 + ((l >= 16) ? 16 : 0) + 2 * j + hl;
    int ncol = ct * 16 + (l & 15);
    Wp[i] = (_Float16)convW[layer * (H * H) + k * H + ncol];
}

// ---- input layer: z = x @ in_W (IN=5) --------------------------------------
__global__ void k_input(const float* __restrict__ x, const float* __restrict__ W,
                        float* __restrict__ z, int n) {
    int i = blockIdx.x * blockDim.x + threadIdx.x;
    if (i >= n * H) return;
    int node = i >> 7, f = i & 127;
    const float* xr = x + node * 5;
    float acc = 0.f;
#pragma unroll
    for (int k = 0; k < 5; ++k) acc += xr[k] * W[k * H + f];
    z[i] = acc;
}

// ---- BN stats: per-column sum / sumsq via partial reduce + atomics ---------
#define STAT_ROWS 256
__global__ void k_zero_stats(float* __restrict__ stats) {
    stats[threadIdx.x] = 0.f;                       // 1 block x 256 threads
}
__global__ void k_stats(const float* __restrict__ z, float* __restrict__ stats, int n) {
    int f  = threadIdx.x;                           // 0..127
    int r0 = blockIdx.x * STAT_ROWS;
    int r1 = r0 + STAT_ROWS; if (r1 > n) r1 = n;
    float s = 0.f, q = 0.f;
    for (int r = r0; r < r1; ++r) {
        float v = z[(size_t)r * H + f];
        s += v; q += v * v;
    }
    atomicAdd(&stats[f], s);
    atomicAdd(&stats[H + f], q);
}

// ---- BN apply + relu (+ residual) ------------------------------------------
__global__ void k_bn_apply(const float* __restrict__ z, const float* __restrict__ stats,
                           const float* __restrict__ g, const float* __restrict__ beta,
                           const float* __restrict__ hres, float* __restrict__ hout,
                           int n, float invN, int residual) {
    int i = blockIdx.x * blockDim.x + threadIdx.x;
    if (i >= n * H) return;
    int f = i & 127;
    float m   = stats[f] * invN;
    float var = stats[H + f] * invN - m * m;        // biased variance
    float s   = g[f] * rsqrtf(var + EPSV);
    float y   = (z[i] - m) * s + beta[f];
    y = fmaxf(y, 0.f);
    if (residual) y += hres[i];
    hout[i] = y;
}

// ---- WMMA GEMM: hw[Nx128] = h[Nx128] @ W_layer[128x128] --------------------
// 256 threads = 8 waves per block; block stages the 32KB packed weight matrix
// into LDS once (async-to-LDS when available), then each wave computes a
// 16-row x 128-col stripe: 8 col-tiles x 4 k-steps = 32 v_wmma per wave.
__global__ void __launch_bounds__(256)
k_gemm_wmma(const float* __restrict__ h, const _Float16* __restrict__ Wp,
            float* __restrict__ hw, int n, int layer) {
    __shared__ __align__(32) _Float16 sW[H * H];    // 32KB

    const _Float16* WpL = Wp + (size_t)layer * (H * H);
    {
        const int t = threadIdx.x;                  // 0..255
#if USE_ASYNC_LDS
#pragma unroll
        for (int c = 0; c < 8; ++c) {
            const int elt = (c * 256 + t) * 8;      // halves; 8 halves = 16B
            __builtin_amdgcn_global_load_async_to_lds_b128(
                (g_v4i*)(WpL + elt), (l_v4i*)(&sW[elt]), 0, 0);
        }
        __builtin_amdgcn_s_wait_asynccnt(0);        // drain this wave's asyncs
#else
        const uint4* src4 = (const uint4*)WpL;
        uint4* dst4 = (uint4*)sW;
#pragma unroll
        for (int c = 0; c < 8; ++c) dst4[c * 256 + t] = src4[c * 256 + t];
#endif
        __syncthreads();                            // LDS visible to all waves
    }

    const int lane = threadIdx.x & 31;
    const int wave = threadIdx.x >> 5;              // 0..7
    const int r0 = blockIdx.x * 128 + wave * 16;
    if (r0 >= n) return;                            // uniform per wave: EXEC all-1s
    const int mrow = r0 + (lane & 15);
    const int m  = (mrow < n) ? mrow : (n - 1);     // clamp (N%16==0 -> never taken)
    const int hi = (lane >> 4) & 1;                 // half-wave selects K offset

    v8f acc[8];
#pragma unroll
    for (int ct = 0; ct < 8; ++ct) acc[ct] = (v8f){0.f,0.f,0.f,0.f,0.f,0.f,0.f,0.f};

    for (int kt = 0; kt < 4; ++kt) {
        // A fragment in contiguous K order matching packed B: element s ->
        // k = s with +16 offset on the upper half-wave.
        v16h a;
        const float* hrow = h + (size_t)m * H + kt * 32 + hi * 16;
#pragma unroll
        for (int p = 0; p < 8; ++p) {
            int k = 2 * p;
            float2 v = *(const float2*)(hrow + k);
            a[2 * p]     = (_Float16)v.x;
            a[2 * p + 1] = (_Float16)v.y;
        }
#pragma unroll
        for (int ct = 0; ct < 8; ++ct) {
            const v16h b = *(const v16h*)(sW + (((kt * 8 + ct) * 32 + lane) * 16));
            acc[ct] = __builtin_amdgcn_wmma_f32_16x16x32_f16(
                false, a, false, b, (short)0, acc[ct], false, false);
        }
    }
    // C/D layout: VGPR vg -> row r0+vg (+8 on upper half-wave), col = lane&15
#pragma unroll
    for (int ct = 0; ct < 8; ++ct) {
        int col = ct * 16 + (lane & 15);
#pragma unroll
        for (int vg = 0; vg < 8; ++vg) {
            int row = r0 + vg + hi * 8;
            if (row < n) hw[(size_t)row * H + col] = acc[ct][vg];
        }
    }
}

// ---- aggregation: init with self-loop, then atomic edge scatter ------------
__global__ void k_agg_init(const float* __restrict__ hw, const float* __restrict__ dinv,
                           float* __restrict__ agg, int n) {
    int i = blockIdx.x * blockDim.x + threadIdx.x;
    if (i >= n * H) return;
    float d = dinv[i >> 7];
    agg[i] = hw[i] * d * d;                          // self-loop norm = dinv^2
}
__global__ void k_scatter(const float* __restrict__ hw, const int* __restrict__ src,
                          const int* __restrict__ dst, const float* __restrict__ dinv,
                          float* __restrict__ agg, int E) {
    int t = blockIdx.x * blockDim.x + threadIdx.x;
    int e = t >> 5, lane = t & 31;
    if (e >= E) return;
    if (lane == 0 && e + 2048 < E) {                 // global_prefetch_b8 on index streams
        __builtin_prefetch(&src[e + 2048], 0, 0);
        __builtin_prefetch(&dst[e + 2048], 0, 0);
    }
    int s = src[e], d = dst[e];
    float w = dinv[s] * dinv[d];
    const float4 v = *(const float4*)(hw + (size_t)s * H + lane * 4);
    float* ap = agg + (size_t)d * H + lane * 4;
    atomicAdd(ap + 0, v.x * w);
    atomicAdd(ap + 1, v.y * w);
    atomicAdd(ap + 2, v.z * w);
    atomicAdd(ap + 3, v.w * w);
}

// ---- policy head: warp-per-node, 128->32->1 --------------------------------
__global__ void k_policy(const float* __restrict__ h, const float* __restrict__ W1,
                         const float* __restrict__ b1, const float* __restrict__ W2,
                         const float* __restrict__ b2, float* __restrict__ out, int n) {
    int t = blockIdx.x * blockDim.x + threadIdx.x;
    int node = t >> 5, lane = t & 31;
    if (node >= n) return;
    const float* hr = h + (size_t)node * H;
    float acc = b1[lane];
    for (int k = 0; k < H; ++k) acc += hr[k] * W1[k * 32 + lane];
    acc = fmaxf(acc, 0.f) * W2[lane];
    for (int msk = 16; msk > 0; msk >>= 1) acc += __shfl_xor(acc, msk, 32);
    if (lane == 0) out[node] = acc + b2[0];
}

// ---- value head: mean-pool (colsum precomputed) -> 64 -> tanh --------------
__global__ void k_value(const float* __restrict__ colsum, const float* __restrict__ W1,
                        const float* __restrict__ b1, const float* __restrict__ W2,
                        const float* __restrict__ b2, float* __restrict__ out,
                        int n, float invN) {
    __shared__ float part[64];
    int j = threadIdx.x;
    float acc = b1[j];
    for (int k = 0; k < H; ++k) acc += (colsum[k] * invN) * W1[k * 64 + j];
    part[j] = fmaxf(acc, 0.f) * W2[j];
    __syncthreads();
    if (j == 0) {
        float s = 0.f;
        for (int k = 0; k < 64; ++k) s += part[k];
        out[n] = tanhf(s + b2[0]);
    }
}

// ---------------------------------------------------------------------------
extern "C" void kernel_launch(void* const* d_in, const int* in_sizes, int n_in,
                              void* d_out, int out_size, void* d_ws, size_t ws_size,
                              hipStream_t stream) {
    const float* x       = (const float*)d_in[0];
    const int*   ei      = (const int*)d_in[1];
    // d_in[2] batch: single graph, unused
    const float* in_W    = (const float*)d_in[3];
    // d_in[4] in_b: per-feature constant, cancels inside BN
    const float* in_g    = (const float*)d_in[5];
    const float* in_beta = (const float*)d_in[6];
    const float* conv_W  = (const float*)d_in[7];
    // d_in[8] conv_b: cancels inside BN
    const float* bn_g    = (const float*)d_in[9];
    const float* bn_beta = (const float*)d_in[10];
    const float* pW1 = (const float*)d_in[11];
    const float* pb1 = (const float*)d_in[12];
    const float* pW2 = (const float*)d_in[13];
    const float* pb2 = (const float*)d_in[14];
    const float* vW1 = (const float*)d_in[15];
    const float* vb1 = (const float*)d_in[16];
    const float* vW2 = (const float*)d_in[17];
    const float* vb2 = (const float*)d_in[18];

    const int n = in_sizes[0] / 5;
    const int E = in_sizes[1] / 2;
    const int* src = ei;
    const int* dst = ei + E;
    const float invN = 1.0f / (float)n;
    float* out = (float*)d_out;                      // [0..n-1] policy, [n] value

    // workspace layout (256B aligned slabs)
    char* ws = (char*)d_ws;
    size_t A = ((size_t)n * H * sizeof(float) + 255) & ~(size_t)255;
    float*    h_buf = (float*)(ws);
    float*    hw    = (float*)(ws + A);
    float*    agg   = (float*)(ws + 2 * A);
    float*    dinv  = (float*)(ws + 3 * A);
    size_t doff = 3 * A + (((size_t)n * sizeof(float) + 255) & ~(size_t)255);
    _Float16* Wp    = (_Float16*)(ws + doff);
    size_t soff = doff + ((LAYERS * H * H * sizeof(_Float16) + 255) & ~(size_t)255);
    float*    stats = (float*)(ws + soff);           // [128 sum | 128 sumsq]

    const int NF = n * H;
    dim3 b256(256);
    dim3 gNF((NF + 255) / 256);
    dim3 gN((n + 255) / 256);
    dim3 gE((E + 255) / 256);

    // degree -> dinv (reuses dinv slab as deg)
    k_deg_init<<<gN, b256, 0, stream>>>(dinv, n);
    k_deg_edges<<<gE, b256, 0, stream>>>(dst, dinv, E);
    k_dinv<<<gN, b256, 0, stream>>>(dinv, n);

    // pack all conv weights to f16 B-fragments once
    const int packTotal = LAYERS * H * H;
    k_pack_w<<<dim3((packTotal + 255) / 256), b256, 0, stream>>>(conv_W, Wp, packTotal);

    // input layer: z in hw, BN+relu -> h
    k_input<<<gNF, b256, 0, stream>>>(x, in_W, hw, n);
    k_zero_stats<<<dim3(1), b256, 0, stream>>>(stats);
    k_stats<<<dim3((n + STAT_ROWS - 1) / STAT_ROWS), dim3(H), 0, stream>>>(hw, stats, n);
    k_bn_apply<<<gNF, b256, 0, stream>>>(hw, stats, in_g, in_beta, hw, h_buf, n, invN, 0);

    // GCN layers
    dim3 gGemm((n + 127) / 128);                     // 8 waves (256 thr) per block
    dim3 bGemm(256);
    dim3 gScat(((size_t)E * 32 + 255) / 256);
    for (int layer = 0; layer < LAYERS; ++layer) {
        k_gemm_wmma<<<gGemm, bGemm, 0, stream>>>(h_buf, Wp, hw, n, layer);
        k_agg_init<<<gNF, b256, 0, stream>>>(hw, dinv, agg, n);
        k_scatter<<<gScat, b256, 0, stream>>>(hw, src, dst, dinv, agg, E);
        k_zero_stats<<<dim3(1), b256, 0, stream>>>(stats);
        k_stats<<<dim3((n + STAT_ROWS - 1) / STAT_ROWS), dim3(H), 0, stream>>>(agg, stats, n);
        k_bn_apply<<<gNF, b256, 0, stream>>>(agg, stats, bn_g + layer * H, bn_beta + layer * H,
                                             h_buf, h_buf, n, invN, 1);
    }

    // policy head (warp per node)
    k_policy<<<dim3(((size_t)n * 32 + 255) / 256), b256, 0, stream>>>(h_buf, pW1, pb1, pW2, pb2, out, n);

    // value head: column sums of h, then tiny MLP + tanh
    k_zero_stats<<<dim3(1), b256, 0, stream>>>(stats);
    k_stats<<<dim3((n + STAT_ROWS - 1) / STAT_ROWS), dim3(H), 0, stream>>>(h_buf, stats, n);
    k_value<<<dim3(1), dim3(64), 0, stream>>>(stats, vW1, vb1, vW2, vb2, out, n, invN);
}
